// SemNN_7267084664869
// MI455X (gfx1250) — matrix operand order repaired and checked by reading, not compiled, not run on hardware
//
#include <hip/hip_runtime.h>
#include <hip/hip_bf16.h>
#include <math.h>

// ---------------------------------------------------------------------------
// SemNN forward for MI455X (gfx1250, wave32, WMMA).
//   B=4096, S=128, D=512, VOCAB=50000, NUM_LABELS=3
// Stages:
//   1) f32->f16 transposed weight prep (LDS tiled transpose)
//   2) embedding gather + masked sum-pool  (memory-bound stage)
//   3) tanh(A @ W + b) GEMMs via v_wmma_f32_16x16x32_f16
//   4) tiny 512->3 logits projection (f32 out)
// ---------------------------------------------------------------------------

typedef _Float16 v16h __attribute__((ext_vector_type(16)));
typedef _Float16 v8h  __attribute__((ext_vector_type(8)));
typedef _Float16 v4h  __attribute__((ext_vector_type(4)));
typedef float    v8f  __attribute__((ext_vector_type(8)));

#define BATCH 4096
#define SEQ   128
#define DIM   512

// ---------------------------------------------------------------------------
// Tiled transpose + f32->f16 convert:  Wt[n*K + k] = (f16) W[k*N + n]
// Grid: (K/32, N/32), block (32,8).
// ---------------------------------------------------------------------------
__global__ void transpose_convert_kernel(const float* __restrict__ W,
                                         _Float16* __restrict__ Wt,
                                         int K, int N) {
    __shared__ float tile[32][33];          // +1 pad: no LDS bank conflicts
    const int kBase = blockIdx.x * 32;
    const int nBase = blockIdx.y * 32;
    const int tx = threadIdx.x;
    for (int i = threadIdx.y; i < 32; i += 8)
        tile[i][tx] = W[(size_t)(kBase + i) * N + (nBase + tx)];
    __syncthreads();
    for (int i = threadIdx.y; i < 32; i += 8)
        Wt[(size_t)(nBase + i) * K + (kBase + tx)] = (_Float16)tile[tx][i];
}

// ---------------------------------------------------------------------------
// Embedding gather + masked sum-pool. One block per batch row, 128 threads,
// each thread owns 4 consecutive floats of the 512-wide row (float4 loads,
// fully coalesced across the wave). Mask is numpy bool (1 byte).
// Accumulate f32, store f16 for the WMMA GEMM that follows.
// ---------------------------------------------------------------------------
__global__ void pool_kernel(const int* __restrict__ ids,
                            const unsigned char* __restrict__ mask,
                            const float* __restrict__ W_emb,
                            _Float16* __restrict__ out) {
    const int b  = blockIdx.x;
    const int t4 = threadIdx.x * 4;          // 128 thr * 4 = 512 floats
    const int* __restrict__ idRow           = ids  + (size_t)b * SEQ;
    const unsigned char* __restrict__ mRow  = mask + (size_t)b * SEQ;

    float4 acc = make_float4(0.f, 0.f, 0.f, 0.f);
    for (int s = 0; s < SEQ; ++s) {
        const int id = idRow[s];
        const float m = (float)mRow[s];
        if (s + 1 < SEQ)   // global_prefetch_b8 of next gather row
            __builtin_prefetch(W_emb + (size_t)idRow[s + 1] * DIM + t4, 0, 0);
        const float4 e = *(const float4*)(W_emb + (size_t)id * DIM + t4);
        acc.x += m * e.x; acc.y += m * e.y;
        acc.z += m * e.z; acc.w += m * e.w;
    }
    v4h r = { (_Float16)acc.x, (_Float16)acc.y, (_Float16)acc.z, (_Float16)acc.w };
    *(v4h*)(out + (size_t)b * DIM + t4) = r;
}

// ---------------------------------------------------------------------------
// C[m, n] = tanh( sum_k A[m,k] * Wt[n,k] + bias[n] ), f16 in, f32 accum,
// f16 out at row stride ldC (lets encoder layer 2 write straight into the
// concat buffer halves).
//
// A: row-major [M, K] f16.  Wt: [N, K] f16 (pre-transposed weight).
// Block: 256 threads = 8 waves as 4(M) x 2(N); each wave owns a 16x64 tile
// (4 wmma accumulators) -> block tile 64 x 128. M=4096, N in {512}, K in
// {512,1024}: every dim divides evenly, no divergence, EXEC all-ones.
//
// Fragment feeds use contiguous k order (same for A and B):
//   A 16x32 f16: lane m(<16): elems 0-15 = K k0..k0+15;
//                lane m+16: K k0+16..+31.        -> two contiguous v8h loads.
//   B 32x16 f16: lane n(<16): elems 0-15 = column n, K k0..k0+15;
//                lane n+16: K k0+16..+31.        -> one contiguous v16h load.
// ---------------------------------------------------------------------------
__global__ void gemm_bias_tanh_f16_kernel(const _Float16* __restrict__ A,
                                          const _Float16* __restrict__ Wt,
                                          const float* __restrict__ bias,
                                          _Float16* __restrict__ C,
                                          int K, int ldC) {
    const int lane  = threadIdx.x & 31;
    const int wave  = threadIdx.x >> 5;
    const int waveM = wave & 3;
    const int waveN = wave >> 2;

    const int m0 = blockIdx.x * 64  + waveM * 16;
    const int n0 = blockIdx.y * 128 + waveN * 64;

    const int rowA   = m0 + (lane & 15);
    const int hiHalf = (lane & 16) ? 1 : 0;

    const _Float16* __restrict__ aRow = A + (size_t)rowA * K;
    const _Float16* __restrict__ bCol0 = Wt + (size_t)(n0 + (lane & 15)) * K;

    v8f acc[4] = {};

    for (int k0 = 0; k0 < K; k0 += 32) {
        const int kbA = k0 + hiHalf * 16;
        v8h alo = *(const v8h*)(aRow + kbA);
        v8h ahi = *(const v8h*)(aRow + kbA + 8);
        v16h a = __builtin_shufflevector(alo, ahi,
                    0, 1, 2, 3, 4, 5, 6, 7, 8, 9, 10, 11, 12, 13, 14, 15);

        const int kbB = k0 + hiHalf * 16;
#pragma unroll
        for (int t = 0; t < 4; ++t) {
            v16h bfrag = *(const v16h*)(bCol0 + (size_t)(t * 16) * K + kbB);
            acc[t] = __builtin_amdgcn_wmma_f32_16x16x32_f16(
                         false, a, false, bfrag, (short)0, acc[t], false, false);
        }
    }

    // Epilogue. C/D layout: VGPR v -> M = m0 + v + (lane>=16 ? 8 : 0),
    // N = n0 + t*16 + (lane & 15).
    const int rBase = m0 + hiHalf * 8;
    const int cLane = lane & 15;
#pragma unroll
    for (int t = 0; t < 4; ++t) {
        const int col = n0 + t * 16 + cLane;
        const float bv = bias[col];
#pragma unroll
        for (int v = 0; v < 8; ++v) {
            float r = tanhf(acc[t][v] + bv);
            C[(size_t)(rBase + v) * ldC + col] = (_Float16)r;
        }
    }
}

// ---------------------------------------------------------------------------
// logits[b, l] = sum_k h[b,k] * cls_w2[k,l] + cls_b2[l]   (f32 out, tiny)
// ---------------------------------------------------------------------------
__global__ void logits_kernel(const _Float16* __restrict__ h,
                              const float* __restrict__ w2,
                              const float* __restrict__ b2,
                              float* __restrict__ out) {
    const int idx = blockIdx.x * blockDim.x + threadIdx.x;   // [0, 4096*3)
    const int b = idx / 3;
    const int l = idx - b * 3;
    const _Float16* __restrict__ hr = h + (size_t)b * DIM;
    float acc = b2[l];
#pragma unroll 4
    for (int k = 0; k < DIM; ++k)
        acc += (float)hr[k] * w2[k * 3 + l];
    out[idx] = acc;
}

// ---------------------------------------------------------------------------
extern "C" void kernel_launch(void* const* d_in, const int* in_sizes, int n_in,
                              void* d_out, int out_size, void* d_ws, size_t ws_size,
                              hipStream_t stream) {
    (void)in_sizes; (void)n_in; (void)out_size; (void)ws_size;

    const int*           idsA   = (const int*)d_in[0];
    const int*           idsB   = (const int*)d_in[1];
    const unsigned char* maskA  = (const unsigned char*)d_in[2];
    const unsigned char* maskB  = (const unsigned char*)d_in[3];
    const float*         W_emb  = (const float*)d_in[4];
    const float*         enc_w1 = (const float*)d_in[5];
    const float*         enc_b1 = (const float*)d_in[6];
    const float*         enc_w2 = (const float*)d_in[7];
    const float*         enc_b2 = (const float*)d_in[8];
    const float*         cls_w1 = (const float*)d_in[9];
    const float*         cls_b1 = (const float*)d_in[10];
    const float*         cls_w2 = (const float*)d_in[11];
    const float*         cls_b2 = (const float*)d_in[12];
    float*               out    = (float*)d_out;

    // Workspace carve-up (f16 elements), ~26 MB total.
    _Float16* w1t     = (_Float16*)d_ws;                 // [512 ][512 ]
    _Float16* w2t     = w1t     + (size_t)DIM * DIM;     // [512 ][512 ]
    _Float16* cw1t    = w2t     + (size_t)DIM * DIM;     // [512 ][1024]
    _Float16* pooledA = cw1t    + (size_t)DIM * 2 * DIM; // [4096][512 ]
    _Float16* pooledB = pooledA + (size_t)BATCH * DIM;   // [4096][512 ]
    _Float16* x1      = pooledB + (size_t)BATCH * DIM;   // [4096][512 ] (reused)
    _Float16* concat  = x1      + (size_t)BATCH * DIM;   // [4096][1024]
    _Float16* hbuf    = concat  + (size_t)BATCH * 2 * DIM; // [4096][512]

    // 1) Weight prep: transposed f16 copies.
    dim3 tb(32, 8);
    transpose_convert_kernel<<<dim3(DIM / 32, DIM / 32), tb, 0, stream>>>(enc_w1, w1t,  DIM,     DIM);
    transpose_convert_kernel<<<dim3(DIM / 32, DIM / 32), tb, 0, stream>>>(enc_w2, w2t,  DIM,     DIM);
    transpose_convert_kernel<<<dim3(2 * DIM / 32, DIM / 32), tb, 0, stream>>>(cls_w1, cw1t, 2 * DIM, DIM);

    // 2) Gather + masked sum-pool.
    pool_kernel<<<BATCH, 128, 0, stream>>>(idsA, maskA, W_emb, pooledA);
    pool_kernel<<<BATCH, 128, 0, stream>>>(idsB, maskB, W_emb, pooledB);

    // 3) Encoder + classifier GEMMs (WMMA f16 -> f32 accum -> tanh -> f16).
    const dim3 ggrid(BATCH / 64, DIM / 128);   // 64 x 4 blocks, 256 thr each
    // encoder A
    gemm_bias_tanh_f16_kernel<<<ggrid, 256, 0, stream>>>(pooledA, w1t, enc_b1, x1,            DIM,     DIM);
    gemm_bias_tanh_f16_kernel<<<ggrid, 256, 0, stream>>>(x1,      w2t, enc_b2, concat,        DIM, 2 * DIM);
    // encoder B (writes second half of concat)
    gemm_bias_tanh_f16_kernel<<<ggrid, 256, 0, stream>>>(pooledB, w1t, enc_b1, x1,            DIM,     DIM);
    gemm_bias_tanh_f16_kernel<<<ggrid, 256, 0, stream>>>(x1,      w2t, enc_b2, concat + DIM,  DIM, 2 * DIM);
    // classifier hidden layer: K = 1024
    gemm_bias_tanh_f16_kernel<<<ggrid, 256, 0, stream>>>(concat,  cw1t, cls_b1, hbuf,     2 * DIM,     DIM);

    // 4) Final 512 -> 3 projection.
    logits_kernel<<<(BATCH * 3) / 128, 128, 0, stream>>>(hbuf, cls_w2, cls_b2, out);
}
